// BoxHead_32427003085530
// MI455X (gfx1250) — compile-verified, matrix operand order checked by reading, not run on hardware
//
#include <hip/hip_runtime.h>
#include <stdint.h>

typedef __attribute__((ext_vector_type(16))) _Float16 v16h;
typedef __attribute__((ext_vector_type(8)))  _Float16 v8h;
typedef __attribute__((ext_vector_type(4)))  _Float16 v4h;
typedef __attribute__((ext_vector_type(8)))  float    v8f;
typedef __attribute__((ext_vector_type(4)))  float    f32x4;

#define NPROP 16000
#define DHID  1024

// ---------------------------------------------------------------------------
// Async global -> LDS copy (CDNA5), GVS addressing: SGPR64 base + VGPR u32
// byte offset. Tracked by ASYNCcnt; completes LDS-side, no VGPR staging.
// ---------------------------------------------------------------------------
__device__ __forceinline__ void async_copy_b128(const void* sbase, uint32_t goff_bytes,
                                                void* lds_ptr) {
    uint32_t ldsoff = (uint32_t)(uintptr_t)lds_ptr;   // low 32 bits = LDS offset
    asm volatile("global_load_async_to_lds_b128 %0, %1, %2"
                 :
                 : "v"(ldsoff), "v"(goff_bytes), "s"(sbase)
                 : "memory");
}
__device__ __forceinline__ void wait_async0() {
    asm volatile("s_wait_asynccnt 0x0" ::: "memory");
}

// ---------------------------------------------------------------------------
// Fragment load: CDNA5 16-bit A/B layout for V_WMMA_*_16X16X32.
// Lane r (0..15): row/col r, K = {0..7} and {16..23}; lane r+16: K = {8..15},{24..31}
// p = &tile[row_of_lane][k0]; two aligned 16-byte loads.
// ---------------------------------------------------------------------------
__device__ __forceinline__ v16h frag_ld(const _Float16* p) {
    v8h lo = *(const v8h*)p;
    v8h hi = *(const v8h*)(p + 16);
    v16h f;
#pragma unroll
    for (int i = 0; i < 8; ++i) { f[i] = lo[i]; f[i + 8] = hi[i]; }
    return f;
}

__device__ __forceinline__ v8f wmma16x16x32(v16h a, v16h b, v8f c) {
    return __builtin_amdgcn_wmma_f32_16x16x32_f16(false, a, false, b,
                                                  (short)0, c, false, false);
}

// ---------------------------------------------------------------------------
// Transpose + convert: W[K][N] f32 (row-major) -> Wt[N][K] f16 (K-major)
// 32x32 tiles via LDS, coalesced both sides.
// ---------------------------------------------------------------------------
__global__ __launch_bounds__(256)
void transpose_cvt_kernel(const float* __restrict__ W, _Float16* __restrict__ Wt,
                          int K, int N) {
    __shared__ _Float16 s[32][33];
    const int kb = blockIdx.x * 32, nb = blockIdx.y * 32;
    const int tx = threadIdx.x & 31, ty = threadIdx.x >> 5;   // ty 0..7
#pragma unroll
    for (int i = 0; i < 4; ++i) {
        int k = ty + i * 8;
        s[tx][k] = (_Float16)W[(size_t)(kb + k) * N + nb + tx];
    }
    __syncthreads();
#pragma unroll
    for (int i = 0; i < 4; ++i) {
        int n = ty + i * 8;
        Wt[(size_t)(nb + n) * K + kb + tx] = s[n][tx];
    }
}

// ---------------------------------------------------------------------------
// Pack [Wc (1024x4) | Wr (1024x12)] -> Wt[16][1024] f16 (K-major)
// ---------------------------------------------------------------------------
__global__ void pack_heads_kernel(const float* __restrict__ Wc,
                                  const float* __restrict__ Wr,
                                  _Float16* __restrict__ Wt) {
    int idx = blockIdx.x * 256 + threadIdx.x;   // 16384
    int k = idx >> 4;
    int n = idx & 15;
    float v = (n < 4) ? Wc[k * 4 + n] : Wr[k * 12 + (n - 4)];
    Wt[(size_t)n * DHID + k] = (_Float16)v;
}

// ---------------------------------------------------------------------------
// Fused GEMM + bias + ReLU: out_f16[M x 1024] = relu(A[M x K] @ B + bias)
// B given pre-transposed/converted: Bt[N][K] f16 (K-major, L2-resident).
// Tile 128x128, BK=32, 256 threads = 8 waves (4M x 2N), 2x4 WMMA accs/wave.
// Double-buffered LDS; B (and A when f16) staged with async global->LDS,
// A-f32 staged via VGPR + cvt (the single unavoidable HBM stream).
// ---------------------------------------------------------------------------
template <bool A_F16>
__global__ __launch_bounds__(256)
void fused_gemm_relu(const void* __restrict__ Ain,
                     const _Float16* __restrict__ Bt,
                     const float* __restrict__ bias,
                     _Float16* __restrict__ out,
                     int K) {
    constexpr int N = DHID;
    __shared__ __align__(16) _Float16 sA[2][128][40];   // [m][k], padded
    __shared__ __align__(16) _Float16 sB[2][128][40];   // [n][k], padded

    const int t    = threadIdx.x;
    const int lane = t & 31;
    const int w    = t >> 5;
    const int m0   = blockIdx.y * 128;
    const int n0   = blockIdx.x * 128;
    const int wm   = w >> 1;        // 0..3 -> M offset wm*32
    const int wn   = w & 1;         // 0..1 -> N offset wn*64
    const int r    = lane & 15;
    const int half = lane >> 4;
    const int k0   = half * 8;

    const float*    Af = (const float*)Ain;
    const _Float16* Ah = (const _Float16*)Ain;

    v8f acc[2][4];
#pragma unroll
    for (int mi = 0; mi < 2; ++mi)
#pragma unroll
        for (int ni = 0; ni < 4; ++ni) acc[mi][ni] = (v8f)(0.0f);

    f32x4 areg[4];

    // --- staging helpers -------------------------------------------------
    auto stage_B = [&](int kb, int buf) {
#pragma unroll
        for (int i = 0; i < 2; ++i) {
            int c = t + 256 * i;            // 512 x 16B chunks
            int n = c >> 2, kc = (c & 3) * 8;
            uint32_t goff = (uint32_t)(((size_t)(n0 + n) * K + kb + kc) * 2);
            async_copy_b128(Bt, goff, &sB[buf][n][kc]);
        }
    };
    auto stage_A_async = [&](int kb, int buf) {   // A already f16 row-major
#pragma unroll
        for (int i = 0; i < 2; ++i) {
            int c = t + 256 * i;
            int row = c >> 2, kc = (c & 3) * 8;
            uint32_t goff = (uint32_t)(((size_t)(m0 + row) * K + kb + kc) * 2);
            async_copy_b128(Ah, goff, &sA[buf][row][kc]);
        }
    };
    auto load_A_f32 = [&](int kb) {
#pragma unroll
        for (int i = 0; i < 4; ++i) {
            int c = t + 256 * i;            // 1024 x float4 chunks
            int row = c >> 3, kc = (c & 7) * 4;
            areg[i] = *(const f32x4*)(Af + (size_t)(m0 + row) * K + kb + kc);
        }
    };
    auto store_A_cvt = [&](int buf) {
#pragma unroll
        for (int i = 0; i < 4; ++i) {
            int c = t + 256 * i;
            int row = c >> 3, kc = (c & 7) * 4;
            v4h h;
#pragma unroll
            for (int j = 0; j < 4; ++j) h[j] = (_Float16)areg[i][j];
            *(v4h*)&sA[buf][row][kc] = h;
        }
    };
    auto compute = [&](int buf) {
        v16h afr[2], bfr[4];
#pragma unroll
        for (int mi = 0; mi < 2; ++mi)
            afr[mi] = frag_ld(&sA[buf][wm * 32 + mi * 16 + r][k0]);
#pragma unroll
        for (int ni = 0; ni < 4; ++ni)
            bfr[ni] = frag_ld(&sB[buf][wn * 64 + ni * 16 + r][k0]);
#pragma unroll
        for (int mi = 0; mi < 2; ++mi)
#pragma unroll
            for (int ni = 0; ni < 4; ++ni)
                acc[mi][ni] = wmma16x16x32(afr[mi], bfr[ni], acc[mi][ni]);
    };

    // --- prologue: fill buffer 0 -----------------------------------------
    stage_B(0, 0);
    if (A_F16) {
        stage_A_async(0, 0);
    } else {
        load_A_f32(0);
        store_A_cvt(0);
    }
    wait_async0();
    __syncthreads();

    // --- main loop: prefetch next slab, compute current -------------------
    int buf = 0;
    for (int kb = 0; kb < K; kb += 32) {
        const int nxt = buf ^ 1;
        const bool has_next = (kb + 32) < K;
        if (has_next) {
            stage_B(kb + 32, nxt);
            if (A_F16) stage_A_async(kb + 32, nxt);
            else       load_A_f32(kb + 32);
        }
        compute(buf);                       // 8 WMMAs hide the async latency
        if (has_next && !A_F16) store_A_cvt(nxt);
        wait_async0();
        __syncthreads();
        buf = nxt;
    }

    // --- epilogue: bias + ReLU, f16 store ---------------------------------
    // C/D layout: lanes 0-15 VGPR j -> M=j; lanes 16-31 -> M=j+8; col = lane&15
#pragma unroll
    for (int mi = 0; mi < 2; ++mi) {
#pragma unroll
        for (int ni = 0; ni < 4; ++ni) {
            int col  = n0 + wn * 64 + ni * 16 + r;
            int rowb = m0 + wm * 32 + mi * 16 + half * 8;
            float bv = bias[col];
#pragma unroll
            for (int j = 0; j < 8; ++j) {
                float v = acc[mi][ni][j] + bv;
                v = v > 0.0f ? v : 0.0f;
                out[(size_t)(rowb + j) * N + col] = (_Float16)v;
            }
        }
    }
}

// ---------------------------------------------------------------------------
// Heads: [class_logits | box_pred] = H2[16000x1024] @ Wt^T(1024x16) + bias
// One wave per 16-row tile; fragments stream straight from global (16B aligned).
// ---------------------------------------------------------------------------
__global__ __launch_bounds__(32)
void head_kernel(const _Float16* __restrict__ H2,
                 const _Float16* __restrict__ Wt,   // [16][1024] f16
                 const float* __restrict__ bc,
                 const float* __restrict__ br,
                 float* __restrict__ out) {
    const int m0   = blockIdx.x * 16;
    const int lane = threadIdx.x;
    const int r    = lane & 15;
    const int half = lane >> 4;
    const int k0   = half * 8;

    const _Float16* arow = H2 + (size_t)(m0 + r) * DHID + k0;
    const _Float16* brow = Wt + (size_t)r * DHID + k0;

    v8f acc = (v8f)(0.0f);
#pragma unroll 4
    for (int kb = 0; kb < DHID; kb += 32) {
        v16h a = frag_ld(arow + kb);
        v16h b = frag_ld(brow + kb);
        acc = wmma16x16x32(a, b, acc);
    }

    float bv = (r < 4) ? bc[r] : br[r - 4];
#pragma unroll
    for (int j = 0; j < 8; ++j) {
        int row = m0 + j + half * 8;
        float v = acc[j] + bv;
        if (r < 4) out[(size_t)row * 4 + r] = v;                              // logits
        else       out[(size_t)NPROP * 4 + (size_t)row * 12 + (r - 4)] = v;   // boxes
    }
}

// ---------------------------------------------------------------------------
extern "C" void kernel_launch(void* const* d_in, const int* in_sizes, int n_in,
                              void* d_out, int out_size, void* d_ws, size_t ws_size,
                              hipStream_t stream) {
    const float* X  = (const float*)d_in[0];   // 16000 x 12544
    const float* W1 = (const float*)d_in[1];   // 12544 x 1024
    const float* b1 = (const float*)d_in[2];
    const float* W2 = (const float*)d_in[3];   // 1024 x 1024
    const float* b2 = (const float*)d_in[4];
    const float* Wc = (const float*)d_in[5];   // 1024 x 4
    const float* bc = (const float*)d_in[6];
    const float* Wr = (const float*)d_in[7];   // 1024 x 12
    const float* br = (const float*)d_in[8];

    char* ws = (char*)d_ws;
    size_t off = 0;
    _Float16* H1  = (_Float16*)(ws + off); off += (size_t)NPROP * DHID * 2;  // 32.77 MB
    _Float16* H2  = (_Float16*)(ws + off); off += (size_t)NPROP * DHID * 2;  // 32.77 MB
    _Float16* W1t = (_Float16*)(ws + off); off += (size_t)DHID * 12544 * 2;  // 25.69 MB
    _Float16* W2t = (_Float16*)(ws + off); off += (size_t)DHID * DHID * 2;   //  2.10 MB
    _Float16* Wht = (_Float16*)(ws + off); off += (size_t)16 * DHID * 2;     //  32 KB
    float* out = (float*)d_out;

    // prep: weights -> f16, N-major (tiny, L2-resident afterwards)
    transpose_cvt_kernel<<<dim3(12544 / 32, DHID / 32), 256, 0, stream>>>(W1, W1t, 12544, DHID);
    transpose_cvt_kernel<<<dim3(DHID / 32, DHID / 32), 256, 0, stream>>>(W2, W2t, DHID, DHID);
    pack_heads_kernel<<<64, 256, 0, stream>>>(Wc, Wr, Wht);

    // grid.x = N-blocks fastest so the 8 column-blocks of one M-stripe share X in L2
    fused_gemm_relu<false><<<dim3(DHID / 128, NPROP / 128), 256, 0, stream>>>(
        (const void*)X, W1t, b1, H1, 12544);
    fused_gemm_relu<true><<<dim3(DHID / 128, NPROP / 128), 256, 0, stream>>>(
        (const void*)H1, W2t, b2, H2, DHID);

    head_kernel<<<NPROP / 16, 32, 0, stream>>>(H2, Wht, bc, br, out);
}
